// StyleLayer_35467839930429
// MI455X (gfx1250) — matrix operand order repaired and checked by reading, not compiled, not run on hardware
//
#include <hip/hip_runtime.h>
#include <hip/hip_bf16.h>
#include <stdint.h>

#define C_IN   512
#define C_OUT  512
#define NSAMP  16
#define HH     64
#define WW     64
#define WDIM   512

#define LSTR   40   // halves per (row,col) cell in LDS (16B aligned, 20-dword stride)
#define LCOLS  66   // 64 pixels + 2 halo columns

typedef __attribute__((ext_vector_type(16))) __bf16     v16bf;
typedef __attribute__((ext_vector_type(8)))  float      v8f;
typedef __attribute__((ext_vector_type(4)))  unsigned   u32x4;

union BFPack { u32x4 u[2]; v16bf bf; };

__device__ __forceinline__ unsigned short f32_to_bf16(float f) {
  unsigned u = __float_as_uint(f);
  u += 0x7FFFu + ((u >> 16) & 1u);   // round-to-nearest-even
  return (unsigned short)(u >> 16);
}

// ---------------------------------------------------------------- w2[o][i] = sum_k conv_w^2
__global__ void w2_kernel(const float* __restrict__ conv_w, float* __restrict__ w2) {
  int idx = blockIdx.x * blockDim.x + threadIdx.x;     // 0 .. 512*512-1
  const float* p = conv_w + (size_t)idx * 9;
  float s = 0.f;
  #pragma unroll
  for (int t = 0; t < 9; ++t) { float v = p[t]; s += v * v; }
  w2[idx] = s;
}

// ---------------------------------------------------------------- s[n][c], d[n][o]
__global__ void style_kernel(const float* __restrict__ w,
                             const float* __restrict__ aw,
                             const float* __restrict__ ab,
                             const float* __restrict__ w2,
                             float* __restrict__ sOut,
                             float* __restrict__ dOut) {
  __shared__ float s_sh[C_IN];
  int n = blockIdx.x, c = threadIdx.x;
  const float* wr = w + (size_t)n * WDIM;
  const float* ar = aw + (size_t)c * WDIM;
  float acc = 0.f;
  #pragma unroll 4
  for (int k = 0; k < WDIM; ++k) acc += wr[k] * ar[k];
  float sv = acc + ab[c];
  s_sh[c] = sv;
  sOut[n * C_IN + c] = sv;
  __syncthreads();
  const float* w2r = w2 + (size_t)c * C_IN;            // c plays the role of out-channel o
  float a2 = 0.f;
  #pragma unroll 4
  for (int i = 0; i < C_IN; ++i) { float si = s_sh[i]; a2 += si * si * w2r[i]; }
  dOut[n * C_OUT + c] = rsqrtf(a2 + 1e-8f);
}

// ---------------------------------------------------------------- pack conv_w into WMMA A-layout (bf16)
// wA[((tap*16 + chunk)*32 + mtile)*32 + lane][e] ; 16 halves (32B) per lane per tile
__global__ void wtrans_kernel(const float* __restrict__ conv_w, unsigned short* __restrict__ wA) {
  int b = blockIdx.x;                  // = (tap*16 + chunk)*32 + mtile, 0..4607
  int m = b & 31;
  int c = (b >> 5) & 15;
  int t = b >> 9;
  int l = threadIdx.x >> 4;            // lane 0..31
  int e = threadIdx.x & 15;            // half index 0..15
  int g = l >> 4;
  int k = e + 16 * g;                       // 16-bit A-matrix K mapping (contiguous: k = 16*(lane/16) + e)
  int oc = m * 16 + (l & 15);
  int ic = c * 32 + k;
  int kh = t / 3, kw = t % 3;
  float v = conv_w[(((size_t)oc * C_IN + ic) * 3 + kh) * 3 + kw];
  wA[((size_t)b * 32 + l) * 16 + e] = f32_to_bf16(v);
}

// ---------------------------------------------------------------- main implicit-GEMM conv
__global__ __launch_bounds__(512)
void conv_kernel(const float* __restrict__ x,
                 const float* __restrict__ noise,
                 const float* __restrict__ nstr,
                 const float* __restrict__ conv_b,
                 const float* __restrict__ sArr,
                 const float* __restrict__ dArr,
                 const unsigned short* __restrict__ wA,
                 float* __restrict__ out) {
  __shared__ unsigned short x_s[3 * LCOLS * LSTR];     // 15840 B
  int y    = blockIdx.x;
  int n    = blockIdx.y;
  int tid  = threadIdx.x;
  int wave = tid >> 5;
  int lane = tid & 31;
  int lg   = lane >> 4;                // half-wave group
  int lm   = lane & 15;

  // zero SAME-padding halo columns once (col 0 and 65, all rows, ch 0..31)
  for (int i = tid; i < 3 * 2 * 32; i += 512) {
    int ch  = i & 31;
    int rc  = i >> 5;                  // row*2 + side
    int row = rc >> 1;
    int col = (rc & 1) ? (LCOLS - 1) : 0;
    x_s[(row * LCOLS + col) * LSTR + ch] = 0;
  }

  v8f zero = {};
  v8f acc[2][4];
  #pragma unroll
  for (int mi = 0; mi < 2; ++mi)
    #pragma unroll
    for (int p = 0; p < 4; ++p) acc[mi][p] = zero;

  for (int chunk = 0; chunk < 16; ++chunk) {
    int kc = chunk * 32;
    __syncthreads();                                   // LDS reuse fence
    // stage 32 ch x 3 rows x 64 cols, modulated by s, as bf16 (channel-contiguous)
    #pragma unroll
    for (int i = 0; i < 12; ++i) {
      int idx = tid + i * 512;
      int ch  = idx / 192;
      int rem = idx - ch * 192;
      int row = rem >> 6;
      int col = rem & 63;
      int ry  = y + row - 1;
      const float* xp = x + (((size_t)(n * C_IN + kc + ch) * HH + ry) * WW + col);
      float v = ((unsigned)ry < HH) ? *xp : 0.f;
      if (chunk < 15) __builtin_prefetch(xp + 32 * HH * WW, 0, 1);
      float sv = sArr[n * C_IN + kc + ch];
      x_s[(row * LCOLS + (col + 1)) * LSTR + ch] = f32_to_bf16(v * sv);
    }
    __syncthreads();

    #pragma unroll
    for (int tap = 0; tap < 9; ++tap) {
      int kh = tap / 3, kw = tap % 3;
      // A operands: 2 M-tiles of 16x32 bf16, pre-packed per-lane (2x b128 each)
      BFPack A0, A1;
      const u32x4* ap =
          (const u32x4*)(wA + ((((size_t)tap * 16 + chunk) * 32 + wave * 2) * 32 + lane) * 16);
      A0.u[0] = ap[0];  A0.u[1] = ap[1];
      A1.u[0] = ap[64]; A1.u[1] = ap[65];                // next M-tile = +1024B = +64 u32x4

      // preload ALL 4 B tiles into distinct registers: one ds clause + one wait,
      // then 8 back-to-back WMMAs (no per-pair s_wait_dscnt serialization)
      BFPack B[4];
      #pragma unroll
      for (int p = 0; p < 4; ++p) {
        int col = p * 16 + lm + kw;                      // includes +1 halo bias
        int off = (kh * LCOLS + col) * LSTR + lg * 16;   // 16B aligned
        B[p].u[0] = *(const u32x4*)&x_s[off];
        B[p].u[1] = *(const u32x4*)&x_s[off + 8];
      }
      #pragma unroll
      for (int p = 0; p < 4; ++p) {
        acc[0][p] = __builtin_amdgcn_wmma_f32_16x16x32_bf16(
            false, A0.bf, false, B[p].bf, (short)0, acc[0][p], false, false);
        acc[1][p] = __builtin_amdgcn_wmma_f32_16x16x32_bf16(
            false, A1.bf, false, B[p].bf, (short)0, acc[1][p], false, false);
      }
    }
  }

  // epilogue: demod, bias, noise, LeakyReLU(0.2)
  float ns = nstr[0];
  const float* nrow = noise + ((size_t)n * HH + y) * WW;
  #pragma unroll
  for (int mi = 0; mi < 2; ++mi) {
    #pragma unroll
    for (int r = 0; r < 8; ++r) {
      int oc     = wave * 32 + mi * 16 + lg * 8 + r;     // C-layout: M = vgpr + 8*(lane/16)
      float dv   = dArr[n * C_OUT + oc];
      float bias = conv_b[oc];
      float* orow = out + ((size_t)(n * C_OUT + oc) * HH + y) * WW;
      #pragma unroll
      for (int p = 0; p < 4; ++p) {
        int px  = p * 16 + lm;                           // N = lane%16
        float v = acc[mi][p][r] * dv + bias + nrow[px] * ns;
        orow[px] = (v >= 0.f) ? v : 0.2f * v;
      }
    }
  }
}

// ----------------------------------------------------------------
extern "C" void kernel_launch(void* const* d_in, const int* in_sizes, int n_in,
                              void* d_out, int out_size, void* d_ws, size_t ws_size,
                              hipStream_t stream) {
  (void)in_sizes; (void)n_in; (void)out_size; (void)ws_size;
  const float* x         = (const float*)d_in[0];
  const float* w         = (const float*)d_in[1];
  const float* noise     = (const float*)d_in[2];
  const float* affine_w  = (const float*)d_in[3];
  const float* affine_b  = (const float*)d_in[4];
  const float* nstrength = (const float*)d_in[5];
  const float* conv_w    = (const float*)d_in[6];
  const float* conv_b    = (const float*)d_in[7];
  float* out = (float*)d_out;

  // workspace layout: w2 (1 MB) | s (32 KB) | d (32 KB) | wA bf16 (4.7 MB)
  float* w2   = (float*)d_ws;
  float* sArr = w2 + (size_t)C_OUT * C_IN;
  float* dArr = sArr + (size_t)NSAMP * C_IN;
  unsigned short* wA = (unsigned short*)(dArr + (size_t)NSAMP * C_OUT);

  w2_kernel   <<<(C_OUT * C_IN) / 256, 256, 0, stream>>>(conv_w, w2);
  style_kernel<<<NSAMP, C_IN, 0, stream>>>(w, affine_w, affine_b, w2, sArr, dArr);
  wtrans_kernel<<<9 * 16 * 32, 512, 0, stream>>>(conv_w, wA);
  conv_kernel <<<dim3(HH, NSAMP), 512, 0, stream>>>(x, noise, nstrength, conv_b,
                                                    sArr, dArr, wA, out);
}
